// ClusteringLayer_51883204936045
// MI455X (gfx1250) — matrix operand order repaired and checked by reading, not compiled, not run on hardware
//
#include <hip/hip_runtime.h>

// ---------------------------------------------------------------------------
// Nearest-centroid assignment (ClusteringLayer) for MI455X / gfx1250.
//   x: [32768, 512] f32, centers: [4096, 512] f32
//   out = [x ; centers[argmin_k ||c_k||^2 - 2 x.c_k]]
//
// bf16 WMMA GEMM fused with running argmin:
//   * prep: 0.5*||c||^2 (f32) and NEGATED bf16 centers -> d_ws (one pass)
//   * centers = A operand, streamed via async global->LDS, double buffered
//   * x = B operand, transposed once into 128 persistent VGPRs per lane
//   * C seeded with 0.5*||c||^2  =>  acc = 0.5*distance directly
//   * 2 accumulator chains + 1-deep A prefetch: fits <256 VGPRs, partial waits
//   * LDS rows padded to 520 bf16 (1040 B = 16*65) -> conflict-optimal b128s
// ---------------------------------------------------------------------------

#define DD 512
#define KK 4096
#define NN 32768

typedef __bf16 bf16;
typedef __attribute__((ext_vector_type(16))) __bf16 v16bf;
typedef __attribute__((ext_vector_type(8)))  __bf16 v8bf;
typedef __attribute__((ext_vector_type(8)))  float  v8f;
typedef __attribute__((ext_vector_type(4)))  int    v4i;

struct bf4 { bf16 a, b, c, d; };   // 8-byte packed store unit

#define USE_ASYNC 0
#if defined(__has_builtin)
#if __has_builtin(__builtin_amdgcn_global_load_async_to_lds_b128) && \
    __has_builtin(__builtin_amdgcn_s_wait_asynccnt)
#undef USE_ASYNC
#define USE_ASYNC 1
#endif
#endif
#if !USE_ASYNC
#warning "gfx1250 async-to-LDS builtins unavailable: using sync fallback"
#endif

#if USE_ASYNC
typedef __attribute__((address_space(1))) v4i g_v4i;   // global int4
typedef __attribute__((address_space(3))) v4i l_v4i;   // LDS int4

__device__ __forceinline__ void async_cp16(const void* gsrc, void* ldst) {
    __builtin_amdgcn_global_load_async_to_lds_b128(
        (g_v4i*)gsrc, (l_v4i*)ldst, 0, 0);
}
#endif

// ---------------------------------------------------------------------------
// Prep 1: cnormh[k] = 0.5 * ||c_k||^2 in f32 (exact). One wave per center.
// ---------------------------------------------------------------------------
__global__ void center_halfnorm_kernel(const float* __restrict__ centers,
                                       float* __restrict__ cnormh) {
    const int wave = threadIdx.x >> 5;
    const int lane = threadIdx.x & 31;
    const int k = blockIdx.x * 8 + wave;
    const float* c = centers + (long long)k * DD;
    float s = 0.f;
#pragma unroll
    for (int j = 0; j < DD / 32; ++j) { float v = c[lane + j * 32]; s += v * v; }
#pragma unroll
    for (int m = 16; m >= 1; m >>= 1) s += __shfl_xor(s, m, 32);
    if (lane == 0) cnormh[k] = 0.5f * s;
}

// ---------------------------------------------------------------------------
// Prep 2: cneg[k][d] = bf16(-centers[k][d])  (coalesced one-pass convert)
// ---------------------------------------------------------------------------
__global__ void center_negbf16_kernel(const float* __restrict__ centers,
                                      bf16* __restrict__ cneg) {
    const int idx = blockIdx.x * 256 + threadIdx.x;     // float4 index
    const float4 v = ((const float4*)centers)[idx];
    ((bf4*)(cneg + (long long)idx * 4))[0] =
        { (bf16)(-v.x), (bf16)(-v.y), (bf16)(-v.z), (bf16)(-v.w) };
}

// ---------------------------------------------------------------------------
// Main kernel: fused GEMM + argmin + gather.
// grid: NN/128 = 256 WGs x 256 threads (8 waves, 16 x-rows each)
// ---------------------------------------------------------------------------
#define MBLK 128
#define NT   64
#define XS_STRIDE 520                       // bf16 elems; 1040 B = 16*65
#define CS_STRIDE 520
#define CS_BUF_BYTES (NT * CS_STRIDE * 2)   // 66560
#define REGION_BYTES (2 * CS_BUF_BYTES)     // 133120 (>= MBLK*XS_STRIDE*2)
#define SMEM_CN_OFF  REGION_BYTES
#define SMEM_IDX_OFF (REGION_BYTES + 2 * NT * 4)
#define SMEM_BYTES   (SMEM_IDX_OFF + MBLK * 4)

#define NTILES (KK / NT)                    // 64
#define TILE_CHUNKS 16                      // 16B chunks per thread per tile

__launch_bounds__(256, 1)
__global__ void cluster_argmin_kernel(const float* __restrict__ x,
                                      const float* __restrict__ centers,
                                      const float* __restrict__ cnormh,
                                      const bf16*  __restrict__ cneg,
                                      float* __restrict__ out) {
    __shared__ __align__(16) unsigned char smem[SMEM_BYTES];
    bf16*  xs    = (bf16*)smem;                         // phase 0/1 (overlaid)
    float* cnb   = (float*)(smem + SMEM_CN_OFF);        // [2][NT] half-norms
    int*   bidx  = (int*)(smem + SMEM_IDX_OFF);

    const int tid  = threadIdx.x;
    const int wave = tid >> 5;
    const int lane = tid & 31;
    const int hf   = lane >> 4;
    const int l16  = lane & 15;
    const long long row0 = (long long)blockIdx.x * MBLK;

    float* out_x = out;
    float* out_y = out + (long long)NN * DD;

    // ---- Phase 0: stream x block -> out_x (fused copy) + bf16 LDS tile ----
    {
        const float4* xin  = (const float4*)(x + row0 * DD);
        float4*       xout = (float4*)(out_x + row0 * DD);
#pragma unroll
        for (int it = 0; it < (MBLK * DD / 4) / 256; ++it) {
            const int fidx = it * 256 + tid;
            const float4 v = xin[fidx];
            xout[fidx] = v;
            const int r  = fidx >> 7;
            const int dq = fidx & 127;
            ((bf4*)(xs + r * XS_STRIDE + dq * 4))[0] =
                { (bf16)v.x, (bf16)v.y, (bf16)v.z, (bf16)v.w };
        }
    }
    __syncthreads();

    // ---- Phase 1: one-time transpose into persistent B fragments.
    //      Padded stride => row step = 4 banks => conflict-free gather. ----
    v16bf bfrag[16];
    {
        const int kloc = (wave * 16 + l16) * XS_STRIDE + hf * 8;
#pragma unroll
        for (int kk = 0; kk < 16; ++kk) {
            const int d = kk * 32 + kloc;
            v16bf bv;
#pragma unroll
            for (int j = 0; j < 16; ++j)
                bv[j] = xs[d + (j < 8 ? j : j + 8)];
            bfrag[kk] = bv;
        }
    }
    __syncthreads();    // region now free for center tiles

    float bestVal = 3.4e38f;
    int   bestIdx = 0;

    // tile loader: fully asynchronous (centers tile + half-norm tile)
    auto issue_tile = [&](int t, int buf) {
        bf16* csb = (bf16*)(smem + buf * CS_BUF_BYTES);
        const bf16* src = cneg + (long long)t * NT * DD;
#pragma unroll
        for (int i = 0; i < TILE_CHUNKS; ++i) {
            const int chunk = i * 256 + tid;       // 4096 chunks = 64 KB
            const int n  = chunk >> 6;             // center row in tile
            const int dq = (chunk & 63) * 8;       // bf16 offset in row
#if USE_ASYNC
            async_cp16(src + chunk * 8, csb + n * CS_STRIDE + dq);
#else
            *(v8bf*)(csb + n * CS_STRIDE + dq) = *(const v8bf*)(src + chunk * 8);
#endif
        }
        if (tid < NT / 4) {                         // 16 x 16B = 256 B norms
#if USE_ASYNC
            async_cp16(cnormh + t * NT + tid * 4, cnb + buf * NT + tid * 4);
#else
            *(float4*)(cnb + buf * NT + tid * 4) =
                *(const float4*)(cnormh + t * NT + tid * 4);
#endif
        }
    };

    issue_tile(0, 0);

    // ---- Phase 2: sweep all K centers, double-buffered tiles ----
    for (int t = 0; t < NTILES; ++t) {
        const int cur = t & 1;
#if USE_ASYNC
        __builtin_amdgcn_s_wait_asynccnt(0);   // my tile-t chunks arrived
#endif
        __syncthreads();                        // all arrived; prev compute done
        if (t + 1 < NTILES) issue_tile(t + 1, cur ^ 1);

        const bf16*  cs = (const bf16*)(smem + cur * CS_BUF_BYTES);
        const float* cn = cnb + cur * NT;
        const int koff = hf * 8;

        // 2 independent accumulator chains per pass (2 passes over the tile)
#pragma unroll
        for (int sp = 0; sp < 2; ++sp) {
            const int m0 = sp * 32;                       // subtiles m0, m0+16
            union accu { v8f v; float f[8]; } acc0, acc1;
#pragma unroll
            for (int i = 0; i < 8; ++i) {                 // seed 0.5*||c||^2
                acc0.f[i] = cn[m0 + hf * 8 + i];
                acc1.f[i] = cn[m0 + 16 + hf * 8 + i];
            }
            const bf16* ar0 = cs + (m0 + l16) * CS_STRIDE;
            const bf16* ar1 = cs + (m0 + 16 + l16) * CS_STRIDE;

            union af_t { v16bf v; v8bf h[2]; } a0c, a1c, a0n, a1n;
            a0c.h[0] = *(const v8bf*)(ar0 + koff);
            a0c.h[1] = *(const v8bf*)(ar0 + koff + 16);
            a1c.h[0] = *(const v8bf*)(ar1 + koff);
            a1c.h[1] = *(const v8bf*)(ar1 + koff + 16);
#pragma unroll
            for (int kk = 0; kk < 16; ++kk) {             // acc += (-c)*x
                if (kk < 15) {                            // 1-deep prefetch
                    const int o = (kk + 1) * 32 + koff;
                    a0n.h[0] = *(const v8bf*)(ar0 + o);
                    a0n.h[1] = *(const v8bf*)(ar0 + o + 16);
                    a1n.h[0] = *(const v8bf*)(ar1 + o);
                    a1n.h[1] = *(const v8bf*)(ar1 + o + 16);
                }
                acc0.v = __builtin_amdgcn_wmma_f32_16x16x32_bf16(
                    false, a0c.v, false, bfrag[kk], (short)0, acc0.v,
                    false, false);
                acc1.v = __builtin_amdgcn_wmma_f32_16x16x32_bf16(
                    false, a1c.v, false, bfrag[kk], (short)0, acc1.v,
                    false, false);
                if (kk < 15) { a0c = a0n; a1c = a1n; }
            }

#pragma unroll
            for (int i = 0; i < 8; ++i) {                 // running argmin
                const int   ci0 = t * NT + m0 + hf * 8 + i;
                const float v0  = acc0.f[i];
                if (v0 < bestVal || (v0 == bestVal && ci0 < bestIdx)) {
                    bestVal = v0; bestIdx = ci0;
                }
                const int   ci1 = t * NT + m0 + 16 + hf * 8 + i;
                const float v1  = acc1.f[i];
                if (v1 < bestVal || (v1 == bestVal && ci1 < bestIdx)) {
                    bestVal = v1; bestIdx = ci1;
                }
            }
        }
    }

    // ---- Phase 3: combine the two lane halves (disjoint center classes) ----
    {
        const float ov = __shfl_xor(bestVal, 16, 32);
        const int   oi = __shfl_xor(bestIdx, 16, 32);
        if (ov < bestVal || (ov == bestVal && oi < bestIdx)) {
            bestVal = ov; bestIdx = oi;
        }
        if (hf == 0) bidx[wave * 16 + l16] = bestIdx;
    }
    __syncthreads();

    // ---- Phase 4: gather winning centers -> out_y (L2-resident source) ----
    {
#pragma unroll
        for (int it = 0; it < (MBLK * DD / 4) / 256; ++it) {
            const int fidx = it * 256 + tid;
            const int r  = fidx >> 7;
            const int dq = fidx & 127;
            const int ci = bidx[r];
            const float4 v = ((const float4*)(centers + (long long)ci * DD))[dq];
            ((float4*)(out_y + (row0 + r) * DD))[dq] = v;
        }
    }
}

// ---------------------------------------------------------------------------
extern "C" void kernel_launch(void* const* d_in, const int* in_sizes, int n_in,
                              void* d_out, int out_size, void* d_ws, size_t ws_size,
                              hipStream_t stream) {
    (void)in_sizes; (void)n_in; (void)out_size; (void)ws_size;
    const float* x       = (const float*)d_in[0];
    const float* centers = (const float*)d_in[1];
    float*       out     = (float*)d_out;
    float*       cnormh  = (float*)d_ws;                         // 16 KB
    bf16*        cneg    = (bf16*)((char*)d_ws + KK * 4);        // 4 MB bf16

    center_halfnorm_kernel<<<KK / 8, 256, 0, stream>>>(centers, cnormh);
    center_negbf16_kernel<<<(KK * DD / 4) / 256, 256, 0, stream>>>(centers, cneg);
    cluster_argmin_kernel<<<NN / MBLK, 256, 0, stream>>>(x, centers, cnormh,
                                                         cneg, out);
}
